// SHDDelaysV2_15831249453642
// MI455X (gfx1250) — matrix operand order repaired and checked by reading, not compiled, not run on hardware
//
#include <hip/hip_runtime.h>

typedef __attribute__((ext_vector_type(16))) _Float16 v16h;
typedef __attribute__((ext_vector_type(8)))  _Float16 v8h;
typedef __attribute__((ext_vector_type(8)))  float    v8f;

#define BDIM 256
#define TDIM 250
#define CDIM 700
#define CPAD 704          // per-tap padded channel dim (22 chunks of 32)
#define NTAP 4
#define HDIM 1536
#define ODIM 20
#define OPAD 32
#define FDIM 2800         // C * N_TAPS (logical)
#define FPAD (NTAP*CPAD)  // 2816, padded K for the input GEMM
#define NCH_IN (FPAD/32)  // 88 K-chunks
#define NCH_H  (HDIM/32)  // 48 K-chunks

// ---------------------------------------------------------------------------
// WMMA helpers (gfx1250, wave32). Fragment layouts per CDNA5 ISA 7.12.2:
//  A (16x32 f16): lane L<16 row M=L holds K=0..7 (halves 0-7) and K=16..23
//                 (halves 8-15); lanes 16-31 hold K=8..15 / K=24..31.
//  B (32x16 f16): lane L<16 col N=L holds K=0..7 / K=16..23; lanes 16-31 hold K=8..15 / K=24..31.
//  C/D (16x16 f32): VGPR r: lanes 0-15 -> M=r, N=lane; lanes 16-31 -> M=r+8.
// ---------------------------------------------------------------------------

__device__ inline v8f wmma_f16(v16h a, v16h b, v8f c) {
  return __builtin_amdgcn_wmma_f32_16x16x32_f16(
      /*neg_a=*/false, a, /*neg_b=*/false, b,
      /*c_mod=*/(short)0, c, /*reuse_a=*/false, /*reuse_b=*/false);
}

// Async global->LDS copy, 16 bytes per active lane (CDNA5 §15.18.3 op 98,
// tracked by ASYNCcnt). VDST supplies the per-lane LDS byte address; the low
// 32 bits of a flat shared-aperture pointer are exactly that offset.
__device__ inline void async_load_b128(_Float16* lds_dst, const _Float16* gsrc) {
  asm volatile("global_load_async_to_lds_b128 %0, %1, off"
               :: "v"((unsigned)(unsigned long long)lds_dst),
                  "v"((unsigned long long)gsrc)
               : "memory");
}
__device__ inline void async_wait0() {
  asm volatile("s_wait_asynccnt 0x0" ::: "memory");
}

// A tile staged in LDS row-major: [rows][32] halves (row stride 32 halves).
__device__ inline v16h frag_a(const _Float16* sA, int row_base) {
  int lane = threadIdx.x & 31;
  int m    = row_base + (lane & 15);
  int khi  = (lane & 16) ? 8 : 0;
  v8h lo = *(const v8h*)(sA + m * 32 + khi);        // K = khi .. khi+7
  v8h hi = *(const v8h*)(sA + m * 32 + khi + 16);   // K = khi+16 .. khi+23
  v16h r;
#pragma unroll
  for (int e = 0; e < 8; ++e) { r[e] = lo[e]; r[e + 8] = hi[e]; }
  return r;
}

// B tile staged in LDS *transposed*: [ncols][32] halves, element [n][k].
__device__ inline v16h frag_b(const _Float16* sBT, int col_base) {
  int lane = threadIdx.x & 31;
  int n    = col_base + (lane & 15);
  int k0   = (lane & 16) ? 8 : 0;
  v8h lo = *(const v8h*)(sBT + n * 32 + k0);        // K = k0 .. k0+7
  v8h hi = *(const v8h*)(sBT + n * 32 + k0 + 16);   // K = k0+16 .. k0+23
  v16h r;
#pragma unroll
  for (int e = 0; e < 8; ++e) { r[e] = lo[e]; r[e + 8] = hi[e]; }
  return r;
}

// ---------------------------------------------------------------------------
// Prep kernels: f32 -> f16 conversions with layout padding.
// ---------------------------------------------------------------------------
__global__ void conv_x16(const float* __restrict__ x, _Float16* __restrict__ O) {
  long idx = (long)blockIdx.x * 256 + threadIdx.x;   // over B*T*CPAD
  if (idx >= (long)BDIM * TDIM * CPAD) return;
  long m = idx / CPAD;                 // b*T + t
  int  c = (int)(idx - m * CPAD);
  O[idx] = (c < CDIM) ? (_Float16)x[m * CDIM + c] : (_Float16)0.f;
}

// Wd16[h][tap*704 + c] = W_delay[h][tap*700 + c] (zeros for c in [700,704))
__global__ void conv_wdelay(const float* __restrict__ W, _Float16* __restrict__ O) {
  long idx = (long)blockIdx.x * 256 + threadIdx.x;   // over HDIM*FPAD
  if (idx >= (long)HDIM * FPAD) return;
  int h = (int)(idx / FPAD);
  int f = (int)(idx - (long)h * FPAD);
  int tap = f / CPAD, c = f - tap * CPAD;
  O[idx] = (c < CDIM) ? (_Float16)W[(long)h * FDIM + tap * CDIM + c]
                      : (_Float16)0.f;
}

__global__ void conv_wrec(const float* __restrict__ W, _Float16* __restrict__ O) {
  long idx = (long)blockIdx.x * 256 + threadIdx.x;
  if (idx >= (long)HDIM * HDIM) return;
  O[idx] = (_Float16)W[idx];
}

__global__ void conv_wout(const float* __restrict__ W, _Float16* __restrict__ O) {
  int idx = blockIdx.x * 256 + threadIdx.x;
  if (idx >= OPAD * HDIM) return;
  int o = idx / HDIM;
  O[idx] = (o < ODIM) ? (_Float16)W[idx] : (_Float16)0.f;
}

// ---------------------------------------------------------------------------
// Kernel 1: I_in[b*T+t][h] = sum_f xd[(b,t),f] * W_delay[h,f]
//   xd[(b,t), tap*704+c] = x16[b][t-10*tap][c]  (delays exact, d_frac==0).
//   Block tile 64(M) x 128(N); 8 waves, each owns 32x32 (2x2 WMMA accums).
//   K chunks of 32; double-buffered LDS; staging via async global->LDS B128
//   (ASYNCcnt), one s_wait_asynccnt + barrier per chunk.
// ---------------------------------------------------------------------------
__global__ __launch_bounds__(256)
void input_gemm(const _Float16* __restrict__ x16,
                const _Float16* __restrict__ Wd16,
                float* __restrict__ Iin) {
  __shared__ _Float16 sA[2][64 * 32];
  __shared__ _Float16 sBT[2][128 * 32];
  const int tid  = threadIdx.x;
  const int wave = tid >> 5;
  const int M0   = blockIdx.y * 64;
  const int N0   = blockIdx.x * 128;
  const int wm   = (wave >> 2) * 32;   // 0 / 32
  const int wn   = (wave & 3) * 32;    // 0,32,64,96

  v8f acc00 = {}, acc01 = {}, acc10 = {}, acc11 = {};

  // Fixed per-thread staging roles
  const int arow  = tid >> 2;            // 0..63
  const int akseg = (tid & 3) * 8;       // 0,8,16,24
  const int am    = M0 + arow;
  const int ab    = am / TDIM;
  const int at    = am - ab * TDIM;
  const _Float16* xbase = x16 + (long)ab * TDIM * CPAD;
  const int bn    = tid >> 1;            // 0..127
  const int bkseg = (tid & 1) * 16;      // 0,16
  const _Float16* brow  = Wd16 + (long)(N0 + bn) * FPAD;

  auto stage = [&](int ch, int buf) {
    int tap = ch / 22;                   // 22 chunks per tap (704/32)
    int ck  = (ch - tap * 22) * 32;
    int ts  = at - 10 * tap;
    _Float16* adst = sA[buf] + arow * 32 + akseg;
    if (ts >= 0) {
      async_load_b128(adst, xbase + (long)ts * CPAD + ck + akseg);
    } else {
      v8h z = {};
      *(v8h*)adst = z;                   // zero-pad causal boundary
    }
    const _Float16* src = brow + tap * CPAD + ck + bkseg;
    _Float16* bdst = sBT[buf] + bn * 32 + bkseg;
    async_load_b128(bdst,     src);
    async_load_b128(bdst + 8, src + 8);
  };

  stage(0, 0);
  for (int ch = 0; ch < NCH_IN; ++ch) {
    const int cur = ch & 1;
    async_wait0();                       // our async writes into LDS[cur] done
    __syncthreads();                     // LDS[cur] ready; LDS[cur^1] free
    if (ch + 1 < NCH_IN) stage(ch + 1, cur ^ 1);
    v16h a0 = frag_a(sA[cur], wm);
    v16h a1 = frag_a(sA[cur], wm + 16);
    v16h b0 = frag_b(sBT[cur], wn);
    v16h b1 = frag_b(sBT[cur], wn + 16);
    acc00 = wmma_f16(a0, b0, acc00);
    acc01 = wmma_f16(a0, b1, acc01);
    acc10 = wmma_f16(a1, b0, acc10);
    acc11 = wmma_f16(a1, b1, acc11);
  }

  const int lane = tid & 31;
  const int mb   = (lane & 16) ? 8 : 0;
  const int nl   = lane & 15;
#pragma unroll
  for (int r = 0; r < 8; ++r) {
    int m0 = M0 + wm + r + mb;
    int m1 = m0 + 16;
    int c0 = N0 + wn + nl;
    int c1 = c0 + 16;
    Iin[(long)m0 * HDIM + c0] = acc00[r];
    Iin[(long)m0 * HDIM + c1] = acc01[r];
    Iin[(long)m1 * HDIM + c0] = acc10[r];
    Iin[(long)m1 * HDIM + c1] = acc11[r];
  }
}

// ---------------------------------------------------------------------------
// Per-timestep recurrent kernel (same pipelined skeleton):
//   I1 = I_in[:,t,:] + spk_prev @ W_rec^T ; LIF update; write v,a,spk_next.
//   M=B=256 (grid.y=4), N=H=1536 (grid.x=12), K=H=1536.
// ---------------------------------------------------------------------------
__global__ __launch_bounds__(256)
void step_kernel(const float* __restrict__ Iin, int t,
                 const _Float16* __restrict__ spk_prev,
                 const _Float16* __restrict__ Wr16,
                 const float* __restrict__ alpha, const float* __restrict__ rho,
                 const float* __restrict__ beta_a,
                 float* __restrict__ v, float* __restrict__ a,
                 _Float16* __restrict__ spk_next) {
  __shared__ _Float16 sA[2][64 * 32];
  __shared__ _Float16 sBT[2][128 * 32];
  const int tid  = threadIdx.x;
  const int wave = tid >> 5;
  const int M0   = blockIdx.y * 64;
  const int N0   = blockIdx.x * 128;
  const int wm   = (wave >> 2) * 32;
  const int wn   = (wave & 3) * 32;

  v8f acc00 = {}, acc01 = {}, acc10 = {}, acc11 = {};

  const int arow  = tid >> 2;
  const int akseg = (tid & 3) * 8;
  const _Float16* asrc = spk_prev + (long)(M0 + arow) * HDIM + akseg;
  const int bn    = tid >> 1;
  const int bkseg = (tid & 1) * 16;
  const _Float16* bsrc = Wr16 + (long)(N0 + bn) * HDIM + bkseg;

  auto stage = [&](int ch, int buf) {
    async_load_b128(sA[buf] + arow * 32 + akseg,    asrc + ch * 32);
    _Float16* bdst = sBT[buf] + bn * 32 + bkseg;
    async_load_b128(bdst,     bsrc + ch * 32);
    async_load_b128(bdst + 8, bsrc + ch * 32 + 8);
  };

  stage(0, 0);
  for (int ch = 0; ch < NCH_H; ++ch) {
    const int cur = ch & 1;
    async_wait0();
    __syncthreads();
    if (ch + 1 < NCH_H) stage(ch + 1, cur ^ 1);
    v16h a0 = frag_a(sA[cur], wm);
    v16h a1 = frag_a(sA[cur], wm + 16);
    v16h b0 = frag_b(sBT[cur], wn);
    v16h b1 = frag_b(sBT[cur], wn + 16);
    acc00 = wmma_f16(a0, b0, acc00);
    acc01 = wmma_f16(a0, b1, acc01);
    acc10 = wmma_f16(a1, b0, acc10);
    acc11 = wmma_f16(a1, b1, acc11);
  }

  const int lane = tid & 31;
  const int mb   = (lane & 16) ? 8 : 0;
  const int nl   = lane & 15;

  auto epilogue = [&](int m, int h, float rec) {
    long  idx = (long)m * HDIM + h;
    float I1  = rec + Iin[((long)m * TDIM + t) * HDIM + h];
    float al  = alpha[h];
    float v1  = v[idx];
    float a1  = a[idx];
    float vn  = al * v1 + (1.f - al) * (I1 - a1);
    float s   = (vn - 1.0f > 0.f) ? 1.f : 0.f;   // spike_fn(v - THRESHOLD)
    vn -= s;                                      // v -= spk * THRESHOLD
    a[idx]        = rho[h] * a1 + beta_a[h] * s;
    v[idx]        = vn;
    spk_next[idx] = (_Float16)s;
  };

#pragma unroll
  for (int r = 0; r < 8; ++r) {
    int m0 = M0 + wm + r + mb, m1 = m0 + 16;
    int h0 = N0 + wn + nl,     h1 = h0 + 16;
    epilogue(m0, h0, acc00[r]);
    epilogue(m0, h1, acc01[r]);
    epilogue(m1, h0, acc10[r]);
    epilogue(m1, h1, acc11[r]);
  }
}

// ---------------------------------------------------------------------------
// Per-timestep readout: I_o = spk @ W_out^T (N padded 20->32), then
//   v_out = beta_out*v_out + (1-beta_out)*I_o ; out_sum += v_out.
//   Block tile 64(M) x 32(N), 8 waves = 4Mx2N, one 16x16 frag each. Grid = 4.
// ---------------------------------------------------------------------------
__global__ __launch_bounds__(256)
void out_step(const _Float16* __restrict__ spk,
              const _Float16* __restrict__ Wo16,
              const float* __restrict__ beta_out,
              float* __restrict__ v_out, float* __restrict__ out_sum) {
  __shared__ _Float16 sA[64 * 32];
  __shared__ _Float16 sBT[32 * 32];
  const int tid  = threadIdx.x;
  const int wave = tid >> 5;
  const int M0   = blockIdx.x * 64;
  const int wm   = (wave >> 1) * 16;   // 0,16,32,48
  const int wn   = (wave & 1) * 16;    // 0,16

  v8f acc = {};
  const int arow  = tid >> 2;
  const int akseg = (tid & 3) * 8;
  const int brow  = tid >> 3;          // 0..31 (n)
  const int bkseg = (tid & 7) * 4;     // 0..28 step 4

  for (int kk = 0; kk < HDIM; kk += 32) {
    {
      const _Float16* src = spk + (long)(M0 + arow) * HDIM + kk + akseg;
      *(v8h*)(sA + arow * 32 + akseg) = *(const v8h*)src;
    }
    {
      const _Float16* src = Wo16 + (long)brow * HDIM + kk + bkseg;
#pragma unroll
      for (int i = 0; i < 4; ++i) sBT[brow * 32 + bkseg + i] = src[i];
    }
    __syncthreads();
    v16h af = frag_a(sA, wm);
    v16h bf = frag_b(sBT, wn);
    acc = wmma_f16(af, bf, acc);
    __syncthreads();
  }

  const int lane = tid & 31;
  const int mb   = (lane & 16) ? 8 : 0;
  const int o    = wn + (lane & 15);
  if (o < ODIM) {
    float bo = beta_out[o];
#pragma unroll
    for (int r = 0; r < 8; ++r) {
      int  m   = M0 + wm + r + mb;
      long idx = (long)m * ODIM + o;
      float vo = bo * v_out[idx] + (1.f - bo) * acc[r];
      v_out[idx]    = vo;
      out_sum[idx] += vo;
    }
  }
}

__global__ void finalize(const float* __restrict__ osum, float* __restrict__ out) {
  int idx = blockIdx.x * 256 + threadIdx.x;
  if (idx < BDIM * ODIM) out[idx] = osum[idx] * (1.0f / (float)TDIM);
}

// ---------------------------------------------------------------------------
extern "C" void kernel_launch(void* const* d_in, const int* in_sizes, int n_in,
                              void* d_out, int out_size, void* d_ws, size_t ws_size,
                              hipStream_t stream) {
  const float* x        = (const float*)d_in[0];
  const float* W_delay  = (const float*)d_in[1];
  const float* W_rec    = (const float*)d_in[2];
  const float* W_out    = (const float*)d_in[3];
  const float* alpha    = (const float*)d_in[4];
  const float* rho      = (const float*)d_in[5];
  const float* beta_a   = (const float*)d_in[6];
  const float* beta_out = (const float*)d_in[7];

  char*  ws  = (char*)d_ws;
  size_t off = 0;
  auto alloc = [&](size_t bytes) -> void* {
    void* p = ws + off;
    off += (bytes + 255) & ~(size_t)255;
    return p;
  };

  float*    Iin  = (float*)   alloc((size_t)BDIM * TDIM * HDIM * 4);  // 393 MB
  _Float16* X16  = (_Float16*)alloc((size_t)BDIM * TDIM * CPAD * 2);  //  90 MB
  _Float16* Wd16 = (_Float16*)alloc((size_t)HDIM * FPAD * 2);
  _Float16* Wr16 = (_Float16*)alloc((size_t)HDIM * HDIM * 2);
  _Float16* Wo16 = (_Float16*)alloc((size_t)OPAD * HDIM * 2);
  float*    v    = (float*)   alloc((size_t)BDIM * HDIM * 4);
  float*    aS   = (float*)   alloc((size_t)BDIM * HDIM * 4);
  _Float16* spk0 = (_Float16*)alloc((size_t)BDIM * HDIM * 2);
  _Float16* spk1 = (_Float16*)alloc((size_t)BDIM * HDIM * 2);
  float*    vout = (float*)   alloc((size_t)BDIM * ODIM * 4);
  float*    osum = (float*)   alloc((size_t)BDIM * ODIM * 4);

  // Zero all recurrent state every call (graph-replay deterministic).
  hipMemsetAsync(v,    0, (size_t)BDIM * HDIM * 4, stream);
  hipMemsetAsync(aS,   0, (size_t)BDIM * HDIM * 4, stream);
  hipMemsetAsync(spk0, 0, (size_t)BDIM * HDIM * 2, stream);
  hipMemsetAsync(spk1, 0, (size_t)BDIM * HDIM * 2, stream);
  hipMemsetAsync(vout, 0, (size_t)BDIM * ODIM * 4, stream);
  hipMemsetAsync(osum, 0, (size_t)BDIM * ODIM * 4, stream);

  {
    long n = (long)BDIM * TDIM * CPAD;
    conv_x16<<<(unsigned)((n + 255) / 256), 256, 0, stream>>>(x, X16);
  }
  {
    long n = (long)HDIM * FPAD;
    conv_wdelay<<<(unsigned)((n + 255) / 256), 256, 0, stream>>>(W_delay, Wd16);
  }
  {
    long n = (long)HDIM * HDIM;
    conv_wrec<<<(unsigned)((n + 255) / 256), 256, 0, stream>>>(W_rec, Wr16);
  }
  conv_wout<<<(OPAD * HDIM + 255) / 256, 256, 0, stream>>>(W_out, Wo16);

  // Big feed-forward GEMM: M = B*T = 64000 (y), N = 1536 (x: 12 panels of 128)
  input_gemm<<<dim3(HDIM / 128, (BDIM * TDIM) / 64), 256, 0, stream>>>(X16, Wd16, Iin);

  for (int t = 0; t < TDIM; ++t) {
    _Float16* sp = (t & 1) ? spk1 : spk0;
    _Float16* sn = (t & 1) ? spk0 : spk1;
    step_kernel<<<dim3(HDIM / 128, BDIM / 64), 256, 0, stream>>>(
        Iin, t, sp, Wr16, alpha, rho, beta_a, v, aS, sn);
    out_step<<<dim3(BDIM / 64), 256, 0, stream>>>(sn, Wo16, beta_out, vout, osum);
  }

  finalize<<<(BDIM * ODIM + 255) / 256, 256, 0, stream>>>(osum, (float*)d_out);
}
